// MultiQueryAttention_3624952398461
// MI455X (gfx1250) — matrix operand order repaired and checked by reading, not compiled, not run on hardware
//
#include <hip/hip_runtime.h>
#include <hip/hip_bf16.h>
#include <math.h>

// Problem constants (from reference)
#define BB   2
#define TT   2048
#define DIMM 2048
#define HH   16
#define HDD  128

typedef __bf16 bf16_t;
typedef __attribute__((ext_vector_type(16))) __bf16 v16bf;
typedef __attribute__((ext_vector_type(8)))  __bf16 v8bf;
typedef __attribute__((ext_vector_type(8)))  float  v8f;

#define WMMA_BF16(a, b, c) \
  __builtin_amdgcn_wmma_f32_16x16x32_bf16(false, (a), false, (b), (short)0, (c), false, false)

// Build a v16bf A/B fragment from two 16-byte chunks (lower 8 elems, upper 8 elems).
__device__ __forceinline__ v16bf frag_from2(const bf16_t* p0, const bf16_t* p1) {
  v8bf a = *(const v8bf*)p0;
  v8bf b = *(const v8bf*)p1;
  v16bf r;
#pragma unroll
  for (int i = 0; i < 8; ++i) { r[i] = a[i]; r[i + 8] = b[i]; }
  return r;
}
__device__ __forceinline__ v16bf frag16(const bf16_t* p) { return frag_from2(p, p + 8); }

__device__ __forceinline__ v8f v8f_zero() {
  v8f z;
#pragma unroll
  for (int i = 0; i < 8; ++i) z[i] = 0.0f;
  return z;
}

// Async global -> LDS copy of one 16B chunk per lane (CDNA5 VFLAT async op,
// tracked by ASYNCcnt). VDST is a per-lane LDS byte address; the low 32 bits
// of a generic LDS pointer are the LDS offset (ISA 10.2 aperture rules).
__device__ __forceinline__ void async_copy_b128(const bf16_t* gsrc, void* ldst) {
  unsigned lds = (unsigned)(unsigned long long)ldst;
  asm volatile("global_load_async_to_lds_b128 %0, %1, off"
               :: "v"(lds), "v"(gsrc) : "memory");
}
__device__ __forceinline__ void wait_async0() {
  asm volatile("s_wait_asynccnt 0" ::: "memory");
}

// ---------------------------------------------------------------------------
// Kernel 1a: f32 -> bf16 elementwise conversion (activations)
// ---------------------------------------------------------------------------
__global__ void f32_to_bf16_kernel(const float* __restrict__ src, bf16_t* __restrict__ dst, int n) {
  int i = blockIdx.x * blockDim.x + threadIdx.x;
  if (i < n) dst[i] = (bf16_t)src[i];
}

// ---------------------------------------------------------------------------
// Kernel 1b: weight transpose + convert: W (K x N, f32) -> Wt (N x K, bf16).
// One-time; strided reads are absorbed by the 192MB L2.
// ---------------------------------------------------------------------------
__global__ void transpose_f32_to_bf16_kernel(const float* __restrict__ W, bf16_t* __restrict__ Wt,
                                             int K, int N) {
  int idx = blockIdx.x * blockDim.x + threadIdx.x;
  if (idx >= K * N) return;
  int n = idx / K;
  int k = idx - n * K;
  Wt[idx] = (bf16_t)W[(size_t)k * N + n];
}

// ---------------------------------------------------------------------------
// Kernel 2: bf16 WMMA GEMM  C(f32, MxN) = A(bf16, MxK row-major) * Bt(bf16, N x K row-major)
// Block tile 128x128, 8 waves (4 along M x 2 along N), each wave 32x64 (2x4 WMMA tiles).
// Bt tile is staged into double-buffered LDS via GLOBAL_LOAD_ASYNC_TO_LDS_B128
// (padded rows: 40 elems = 80B -> conflict-free, 16B aligned), pipelined one
// K-step ahead and synchronized with s_wait_asynccnt + workgroup barrier.
// ---------------------------------------------------------------------------
__global__ __launch_bounds__(256)
void gemm_bf16_kernel(const bf16_t* __restrict__ A, const bf16_t* __restrict__ Bt,
                      float* __restrict__ C, int M, int N, int K) {
  __shared__ bf16_t Bs[2][128][40];
  const int tid  = threadIdx.x;
  const int w    = tid >> 5;
  const int lane = tid & 31;
  const int half = lane >> 4;
  const int r    = lane & 15;
  const int waveM = w & 3;
  const int waveN = w >> 2;
  const int mb = blockIdx.y * 128;
  const int nb = blockIdx.x * 128;
  (void)M;

  v8f acc[2][4];
#pragma unroll
  for (int mt = 0; mt < 2; ++mt)
#pragma unroll
    for (int nt = 0; nt < 4; ++nt) acc[mt][nt] = v8f_zero();

  // Stage one 128(N) x 32(K) tile of Bt into Bs[buf] (512 x 16B chunks, 2/thread).
  auto stage = [&](int buf, int kk) {
#pragma unroll
    for (int it = 0; it < 2; ++it) {
      int c  = tid + it * 256;      // 0..511
      int n  = c >> 2;              // 0..127
      int cc = (c & 3) << 3;        // 0,8,16,24 (elems within the 32-wide K row)
      async_copy_b128(&Bt[(size_t)(nb + n) * K + kk + cc], &Bs[buf][n][cc]);
    }
  };

  int buf = 0;
  stage(0, 0);
  for (int kk = 0; kk < K; kk += 32) {
    wait_async0();        // my async fills for Bs[buf] are done
    __syncthreads();      // everyone's fills are done; everyone's reads of Bs[buf^1] are done
    if (kk + 32 < K) stage(buf ^ 1, kk + 32);  // overlap next fill with compute

    v16bf afrag[2];
#pragma unroll
    for (int mt = 0; mt < 2; ++mt) {
      int row = mb + waveM * 32 + mt * 16 + r;
      const bf16_t* ap = &A[(size_t)row * K + kk + half * 16];
      afrag[mt] = frag16(ap);   // K = half*16+{0..15}
      __builtin_prefetch(ap + 32, 0, 1);     // global_prefetch for next K-step
    }
#pragma unroll
    for (int nt = 0; nt < 4; ++nt) {
      int nl = waveN * 64 + nt * 16 + r;
      v16bf bfrag = frag16(&Bs[buf][nl][half * 16]);  // K = half*16+{0..15}
#pragma unroll
      for (int mt = 0; mt < 2; ++mt)
        acc[mt][nt] = WMMA_BF16(afrag[mt], bfrag, acc[mt][nt]);
    }
    buf ^= 1;
  }

#pragma unroll
  for (int mt = 0; mt < 2; ++mt)
#pragma unroll
    for (int nt = 0; nt < 4; ++nt)
#pragma unroll
      for (int j = 0; j < 8; ++j) {
        int row = mb + waveM * 32 + mt * 16 + j + 8 * half;
        int col = nb + waveN * 64 + nt * 16 + r;
        C[(size_t)row * N + col] = acc[mt][nt][j];
      }
}

// ---------------------------------------------------------------------------
// Kernel 3: RoPE on q and k; emit bf16 q (B,T,DIM), bf16 k (B,T,HD),
// bf16 V^T (B,HD,T) for contiguous PV fragments, and f32 present_k / present_v.
// One thread per (b,t,d) with d in [0,HD).
// ---------------------------------------------------------------------------
__global__ void rope_kernel(const float* __restrict__ qf, const float* __restrict__ kf,
                            const float* __restrict__ vf,
                            bf16_t* __restrict__ qb, bf16_t* __restrict__ kb,
                            bf16_t* __restrict__ vtb,
                            float* __restrict__ presk, float* __restrict__ presv) {
  int idx = blockIdx.x * blockDim.x + threadIdx.x;
  if (idx >= BB * TT * HDD) return;
  int d  = idx & (HDD - 1);
  int bt = idx >> 7;
  int t  = bt & (TT - 1);
  int b  = bt >> 11;

  int fi = d & 63;  // angles are concatenated [ang, ang]
  float invf = __powf(10000.0f, -(float)(2 * fi) / (float)HDD);
  float ang  = (float)t * invf;
  float s, c;
  __sincosf(ang, &s, &c);

#pragma unroll 4
  for (int h = 0; h < HH; ++h) {
    size_t base = (size_t)bt * DIMM + h * HDD;
    float x0 = qf[base + d];
    float xr = (d < 64) ? -qf[base + d + 64] : qf[base + d - 64];
    qb[base + d] = (bf16_t)(x0 * c + xr * s);
  }

  size_t kbase = (size_t)bt * HDD;
  float k0 = kf[kbase + d];
  float kr = (d < 64) ? -kf[kbase + d + 64] : kf[kbase + d - 64];
  float kro = k0 * c + kr * s;
  kb[kbase + d]    = (bf16_t)kro;
  presk[kbase + d] = kro;

  float vv = vf[kbase + d];
  presv[kbase + d] = vv;
  vtb[((size_t)b * HDD + d) * TT + t] = (bf16_t)vv;
}

// ---------------------------------------------------------------------------
// Kernel 4: causal flash attention (multi-query: one shared K/V head).
// One wave per 16-row query tile; KV processed in 32-wide blocks.
// ---------------------------------------------------------------------------
__global__ __launch_bounds__(256)
void mqa_attn_kernel(const bf16_t* __restrict__ qb, const bf16_t* __restrict__ kb,
                     const bf16_t* __restrict__ vtb, bf16_t* __restrict__ attnb) {
  __shared__ bf16_t Ps[8][16][32];   // per-wave P staging (C-layout -> A-layout)
  const int tid  = threadIdx.x;
  const int w    = tid >> 5;
  const int lane = tid & 31;
  const int half = lane >> 4;
  const int r    = lane & 15;

  const int QT = TT / 16;            // 128 query tiles per (b,h)
  int gw    = blockIdx.x * 8 + w;
  int qtile = gw % QT;
  int h     = (gw / QT) % HH;
  int b     = gw / (QT * HH);
  int q0    = qtile * 16;
  const float scale = 0.08838834764831845f;  // 1/sqrt(128)

  // Q fragments for the whole tile (M=16, K=128 in 4 chunks of 32), kept in registers.
  v16bf aq[4];
  {
    size_t qrow = (size_t)(b * TT + q0 + r) * DIMM + (size_t)h * HDD;
#pragma unroll
    for (int kc = 0; kc < 4; ++kc) {
      const bf16_t* p = &qb[qrow + kc * 32 + half * 16];
      aq[kc] = frag16(p);
    }
  }

  v8f acc[8];
#pragma unroll
  for (int dd = 0; dd < 8; ++dd) acc[dd] = v8f_zero();
  float mrow[8], lrow[8];
#pragma unroll
  for (int j = 0; j < 8; ++j) { mrow[j] = -3.0e38f; lrow[j] = 0.0f; }

  const int nkv = q0 + 16;           // causal limit (exclusive)
  for (int k0i = 0; k0i < nkv; k0i += 32) {
    // S = Q * K^T for two 16-wide key tiles
    v8f s0 = v8f_zero(), s1 = v8f_zero();
#pragma unroll
    for (int kc = 0; kc < 4; ++kc) {
      v16bf bk0 = frag16(&kb[(size_t)(b * TT + k0i + r)      * HDD + kc * 32 + half * 16]);
      v16bf bk1 = frag16(&kb[(size_t)(b * TT + k0i + 16 + r) * HDD + kc * 32 + half * 16]);
      s0 = WMMA_BF16(aq[kc], bk0, s0);
      s1 = WMMA_BF16(aq[kc], bk1, s1);
    }

    // Online softmax (row j+8*half lives in VGPR j across the 16 lanes of this half)
#pragma unroll
    for (int j = 0; j < 8; ++j) {
      int row = q0 + j + 8 * half;
      int c0  = k0i + r;
      int c1  = k0i + 16 + r;
      float v0 = s0[j] * scale;
      float v1 = s1[j] * scale;
      if (c0 > row) v0 = -3.0e38f;
      if (c1 > row) v1 = -3.0e38f;
      float rm = fmaxf(v0, v1);
#pragma unroll
      for (int off = 8; off > 0; off >>= 1) rm = fmaxf(rm, __shfl_xor(rm, off, 32));
      float newm = fmaxf(mrow[j], rm);
      float p0 = __expf(v0 - newm);
      float p1 = __expf(v1 - newm);
      float rs = p0 + p1;
#pragma unroll
      for (int off = 8; off > 0; off >>= 1) rs += __shfl_xor(rs, off, 32);
      float fac = __expf(mrow[j] - newm);
      lrow[j] = lrow[j] * fac + rs;
      mrow[j] = newm;
#pragma unroll
      for (int dd = 0; dd < 8; ++dd) acc[dd][j] *= fac;
      Ps[w][j + 8 * half][r]      = (bf16_t)p0;
      Ps[w][j + 8 * half][16 + r] = (bf16_t)p1;
    }
    asm volatile("s_wait_dscnt 0" ::: "memory");  // same-wave DS RAW fence

    // O += P * V  (P: 16x32 A-frag from LDS; V^T rows give contiguous B-frags)
    v16bf pa = frag16(&Ps[w][r][half * 16]);
#pragma unroll
    for (int dd = 0; dd < 8; ++dd) {
      v16bf bv = frag16(&vtb[((size_t)b * HDD + dd * 16 + r) * TT + k0i + half * 16]);
      acc[dd] = WMMA_BF16(pa, bv, acc[dd]);
    }
  }

  // Normalize and store as bf16 in (B,T,DIM) layout for the O-projection GEMM.
#pragma unroll
  for (int j = 0; j < 8; ++j) {
    float inv = 1.0f / lrow[j];
    int row = q0 + j + 8 * half;
    size_t obase = (size_t)(b * TT + row) * DIMM + (size_t)h * HDD;
#pragma unroll
    for (int dd = 0; dd < 8; ++dd)
      attnb[obase + dd * 16 + r] = (bf16_t)(acc[dd][j] * inv);
  }
}

// ---------------------------------------------------------------------------
// Host-side orchestration
// ---------------------------------------------------------------------------
extern "C" void kernel_launch(void* const* d_in, const int* in_sizes, int n_in,
                              void* d_out, int out_size, void* d_ws, size_t ws_size,
                              hipStream_t stream) {
  (void)in_sizes; (void)n_in; (void)out_size; (void)ws_size;
  const float* x  = (const float*)d_in[0];
  const float* Wq = (const float*)d_in[1];
  const float* Wk = (const float*)d_in[2];
  const float* Wv = (const float*)d_in[3];
  const float* Wo = (const float*)d_in[4];

  float* out   = (float*)d_out;                         // (B,T,DIM)
  float* presk = out   + (size_t)BB * TT * DIMM;        // (B,1,T,HD)
  float* presv = presk + (size_t)BB * TT * HDD;         // (B,1,T,HD)

  char* ws = (char*)d_ws;
  size_t off = 0;
  auto alloc = [&](size_t bytes) -> void* {
    void* p = ws + off;
    off += (bytes + 255) & ~(size_t)255;
    return p;
  };
  const size_t NX = (size_t)BB * TT * DIMM;   // 8.4M
  const size_t NK = (size_t)BB * TT * HDD;    // 0.52M
  bf16_t* xb    = (bf16_t*)alloc(NX * 2);
  bf16_t* wqt   = (bf16_t*)alloc((size_t)DIMM * DIMM * 2);  // (N=DIM) x (K=DIM)
  bf16_t* wkt   = (bf16_t*)alloc((size_t)DIMM * HDD * 2);   // (N=HD)  x (K=DIM)
  bf16_t* wvt   = (bf16_t*)alloc((size_t)DIMM * HDD * 2);
  bf16_t* wot   = (bf16_t*)alloc((size_t)DIMM * DIMM * 2);
  float*  qf    = (float*)alloc(NX * 4);
  float*  kf    = (float*)alloc(NK * 4);
  float*  vf    = (float*)alloc(NK * 4);
  bf16_t* qrope = (bf16_t*)alloc(NX * 2);
  bf16_t* krope = (bf16_t*)alloc(NK * 2);
  bf16_t* vtb   = (bf16_t*)alloc(NK * 2);
  bf16_t* attnb = (bf16_t*)alloc(NX * 2);

  // 1) bf16 conversion of x; transpose+convert of weights (B becomes N-major)
  f32_to_bf16_kernel<<<(int)((NX + 255) / 256), 256, 0, stream>>>(x, xb, (int)NX);
  transpose_f32_to_bf16_kernel<<<(DIMM * DIMM + 255) / 256, 256, 0, stream>>>(Wq, wqt, DIMM, DIMM);
  transpose_f32_to_bf16_kernel<<<(DIMM * HDD  + 255) / 256, 256, 0, stream>>>(Wk, wkt, DIMM, HDD);
  transpose_f32_to_bf16_kernel<<<(DIMM * HDD  + 255) / 256, 256, 0, stream>>>(Wv, wvt, DIMM, HDD);
  transpose_f32_to_bf16_kernel<<<(DIMM * DIMM + 255) / 256, 256, 0, stream>>>(Wo, wot, DIMM, DIMM);

  // 2) projections: q = x@Wq (N=2048), k = x@Wk, v = x@Wv (N=128)
  const int M = BB * TT;
  dim3 gq(DIMM / 128, M / 128);
  gemm_bf16_kernel<<<gq, 256, 0, stream>>>(xb, wqt, qf, M, DIMM, DIMM);
  dim3 gkv(HDD / 128, M / 128);
  gemm_bf16_kernel<<<gkv, 256, 0, stream>>>(xb, wkt, kf, M, HDD, DIMM);
  gemm_bf16_kernel<<<gkv, 256, 0, stream>>>(xb, wvt, vf, M, HDD, DIMM);

  // 3) RoPE + layout transforms + present_k/present_v
  rope_kernel<<<(int)((NK + 255) / 256), 256, 0, stream>>>(qf, kf, vf, qrope, krope, vtb,
                                                           presk, presv);

  // 4) causal flash attention (one wave per 16-row q tile; 4096 waves total)
  const int nwaves = BB * HH * (TT / 16);
  mqa_attn_kernel<<<nwaves / 8, 256, 0, stream>>>(qrope, krope, vtb, attnb);

  // 5) output projection
  gemm_bf16_kernel<<<gq, 256, 0, stream>>>(attnb, wot, out, M, DIMM, DIMM);
}
